// CustomLSTMCell_13769665150976
// MI455X (gfx1250) — matrix operand order repaired and checked by reading, not compiled, not run on hardware
//
#include <hip/hip_runtime.h>

// ---------------------------------------------------------------------------
// Dual-branch fused LSTM cell for MI455X (gfx1250, wave32, WMMA).
//   B=4096, I=512, H=1024, K = H+I = 1536.  GEMMs: 103 GFLOP (compute bound).
//   - v_wmma_f32_16x16x32_bf16 for the 8 gate GEMMs (fp32 acc).
//   - global_load_async_to_lds_b128 (ASYNCcnt) double-buffered staging,
//     one workgroup barrier per K-step.
//   - fused bias + sigmoid/tanh + cell update epilogue through LDS.
// Workspace: [0, 24MB)  bf16 hx[2][4096][1536]
//            [24,48MB)  bf16 W [8][1024][1536]  (f,i,c,o light; f,i,c,o temp)
// ---------------------------------------------------------------------------

#define B_   4096
#define I_   512
#define H_   1024
#define KD   1536
#define BM   64
#define BN   32
#define KB   32
#define PADK 40            // LDS row stride in bf16 elems (16B-aligned, padded)
#define A_U16   (2 * BM * PADK)          // 5120 u16  = 10240 B
#define B_U16   (8 * BN * PADK)          // 10240 u16 = 20480 B
#define STG_U16 (A_U16 + B_U16)          // 15360 u16 = 30720 B per stage
#define NIT     (KD / KB)                // 48 K-steps

typedef unsigned short u16;
typedef __attribute__((ext_vector_type(16))) __bf16 v16bf;
typedef __attribute__((ext_vector_type(8)))  float  v8f;

union FragU { v16bf v; uint4 q[2]; };

__device__ __forceinline__ u16 f2bf(float f) {
    unsigned int u = __float_as_uint(f);
    u += 0x7FFFu + ((u >> 16) & 1u);          // round-to-nearest-even
    return (u16)(u >> 16);
}

// ------------------------- pack hx = [h | y] -> bf16 ------------------------
__global__ void pack_hx_kernel(const float* __restrict__ y,
                               const float* __restrict__ hl,
                               const float* __restrict__ ht,
                               u16* __restrict__ hx) {
    long t = (long)blockIdx.x * blockDim.x + threadIdx.x;  // 2*B*(KD/4)
    const long tot = 2L * B_ * (KD / 4);
    if (t >= tot) return;
    int k4  = (int)(t % (KD / 4));
    long r  = t / (KD / 4);
    int b   = (int)(r % B_);
    int br  = (int)(r / B_);
    int k   = k4 * 4;
    float4 v;
    if (k < H_) v = *(const float4*)((br ? ht : hl) + (size_t)b * H_ + k);
    else        v = *(const float4*)(y + (size_t)b * I_ + (k - H_));
    ushort4 o;
    o.x = f2bf(v.x); o.y = f2bf(v.y); o.z = f2bf(v.z); o.w = f2bf(v.w);
    *(ushort4*)(hx + ((size_t)br * B_ + b) * KD + k) = o;
}

// ------------------------- pack 8 weight mats -> bf16 -----------------------
__global__ void pack_w_kernel(const float* __restrict__ w0, const float* __restrict__ w1,
                              const float* __restrict__ w2, const float* __restrict__ w3,
                              const float* __restrict__ w4, const float* __restrict__ w5,
                              const float* __restrict__ w6, const float* __restrict__ w7,
                              u16* __restrict__ Wb) {
    long t = (long)blockIdx.x * blockDim.x + threadIdx.x;  // 8*H*(KD/4)
    const long tot = 8L * H_ * (KD / 4);
    if (t >= tot) return;
    int k4 = (int)(t % (KD / 4));
    long r = t / (KD / 4);
    int row = (int)(r % H_);
    int m   = (int)(r / H_);
    const float* w;
    switch (m) {
        case 0: w = w0; break; case 1: w = w1; break;
        case 2: w = w2; break; case 3: w = w3; break;
        case 4: w = w4; break; case 5: w = w5; break;
        case 6: w = w6; break; default: w = w7; break;
    }
    int k = k4 * 4;
    float4 v = *(const float4*)(w + (size_t)row * KD + k);
    ushort4 o;
    o.x = f2bf(v.x); o.y = f2bf(v.y); o.z = f2bf(v.z); o.w = f2bf(v.w);
    *(ushort4*)(Wb + ((size_t)m * H_ + row) * KD + k) = o;
}

// ----------------------------- fused main kernel ----------------------------
__global__ __launch_bounds__(256)
void lstm_main_kernel(const u16* __restrict__ hx, const u16* __restrict__ Wb,
                      const float* __restrict__ cprev,
                      const float* __restrict__ bF1, const float* __restrict__ bI1,
                      const float* __restrict__ bC1, const float* __restrict__ bO1,
                      const float* __restrict__ bF2, const float* __restrict__ bI2,
                      const float* __restrict__ bC2, const float* __restrict__ bO2,
                      float* __restrict__ outH, float* __restrict__ outC) {
    union SMem {
        u16   stage[2][STG_U16];     // 61440 B, double-buffered async staging
        float z[2][4][32 * 32];      // 32768 B, epilogue scratch (reused)
    };
    __shared__ __align__(16) SMem sm;

    const int tid    = threadIdx.x;
    const int lane   = tid & 31;
    const int wave   = tid >> 5;
    const int br     = wave >> 2;        // branch this wave computes
    const int q      = wave & 3;
    const int mtBase = (q >> 1) * 2;     // first of 2 m-tiles (16 rows each)
    const int ntBase = (q & 1) * 4;      // first of 4 n-tiles (gate-major)
    const int lane15 = lane & 15;
    const int kh     = lane >> 4;        // K-half selector in frag layouts

    const int rowBase = blockIdx.x * BM;
    const int colBase = blockIdx.y * BN;

    v8f acc[2][4];
    #pragma unroll
    for (int i = 0; i < 2; ++i)
        #pragma unroll
        for (int j = 0; j < 4; ++j) acc[i][j] = (v8f)(0.0f);

    // -------- async staging setup: 6 x 16B chunks per thread per K-step -----
    // LDS byte address of workgroup allocation.
    const unsigned smBase = (unsigned)(uintptr_t)&sm;
    unsigned gOff[6];   // global byte offsets (advance +64 B per K-step)
    unsigned lOff[6];   // stage-0 LDS byte addresses
    {
        #pragma unroll
        for (int j = 0; j < 2; ++j) {            // A: 512 chunks, 2/thread
            const int id  = tid * 2 + j;
            const int abr = id >> 8;
            const int row = (id >> 2) & 63;
            const int ch  = id & 3;
            gOff[j] = (unsigned)(((abr * B_ + rowBase + row) * KD + ch * 8) * 2);
            lOff[j] = smBase + (unsigned)(((abr * BM + row) * PADK + ch * 8) * 2);
        }
        #pragma unroll
        for (int j = 0; j < 4; ++j) {            // W: 1024 chunks, 4/thread
            const int id  = tid * 4 + j;
            const int mat = id >> 7;
            const int row = (id >> 2) & 31;
            const int ch  = id & 3;
            gOff[2 + j] = (unsigned)(((mat * H_ + colBase + row) * KD + ch * 8) * 2);
            lOff[2 + j] = smBase + (unsigned)(A_U16 * 2) +
                          (unsigned)(((mat * BN + row) * PADK + ch * 8) * 2);
        }
    }

    // Prologue: fill stage 0 for k0 = 0.
    #pragma unroll
    for (int j = 0; j < 6; ++j) {
        if (j < 2)
            asm volatile("global_load_async_to_lds_b128 %0, %1, %2"
                         :: "v"(lOff[j]), "v"(gOff[j]), "s"(hx) : "memory");
        else
            asm volatile("global_load_async_to_lds_b128 %0, %1, %2"
                         :: "v"(lOff[j]), "v"(gOff[j]), "s"(Wb) : "memory");
        gOff[j] += KB * 2;
    }

    u16* smu = (u16*)&sm;

    for (int it = 0; it < NIT; ++it) {
        const int cur = it & 1;
        // My fills of stage[cur] have landed:
        asm volatile("s_wait_asynccnt 0x0" ::: "memory");
        // Everyone's fills landed AND everyone is done reading stage[cur^1]:
        __syncthreads();
        // Kick off fills of the other buffer (overlaps with compute below).
        if (it + 1 < NIT) {
            const unsigned so = (unsigned)((cur ^ 1) * STG_U16 * 2);
            #pragma unroll
            for (int j = 0; j < 6; ++j) {
                const unsigned la = lOff[j] + so;
                if (j < 2)
                    asm volatile("global_load_async_to_lds_b128 %0, %1, %2"
                                 :: "v"(la), "v"(gOff[j]), "s"(hx) : "memory");
                else
                    asm volatile("global_load_async_to_lds_b128 %0, %1, %2"
                                 :: "v"(la), "v"(gOff[j]), "s"(Wb) : "memory");
                gOff[j] += KB * 2;
            }
        }

        const u16* stA = smu + cur * STG_U16;          // A region of this stage
        const u16* stB = stA + A_U16;                  // W region of this stage

        // A fragments: 16-bit A 16x32 layout (K chunks kh*8 and 16+kh*8)
        FragU a[2];
        #pragma unroll
        for (int mt = 0; mt < 2; ++mt) {
            const u16* p = stA + ((br * BM + (mtBase + mt) * 16 + lane15) * PADK + kh * 8);
            a[mt].q[0] = *(const uint4*)(p);
            a[mt].q[1] = *(const uint4*)(p + 16);
        }
        #pragma unroll
        for (int nt = 0; nt < 4; ++nt) {
            const int ntg = ntBase + nt;
            const int g   = ntg >> 1;
            const int ct  = ntg & 1;
            const int m   = br * 4 + g;
            // B fragment: 16-bit B 32x16 layout (lane=n, K chunks kh*8 and 16+kh*8)
            FragU bf;
            const u16* p = stB + ((m * BN + ct * 16 + lane15) * PADK + kh * 8);
            bf.q[0] = *(const uint4*)(p);
            bf.q[1] = *(const uint4*)(p + 16);
            acc[0][nt] = __builtin_amdgcn_wmma_f32_16x16x32_bf16(
                false, a[0].v, false, bf.v, (short)0, acc[0][nt], false, false);
            acc[1][nt] = __builtin_amdgcn_wmma_f32_16x16x32_bf16(
                false, a[1].v, false, bf.v, (short)0, acc[1][nt], false, false);
        }
    }

    // -------- epilogue: two 32-row passes through LDS, fused LSTM math ------
    for (int pass = 0; pass < 2; ++pass) {
        __syncthreads();
        if ((q >> 1) == pass) {
            #pragma unroll
            for (int mt = 0; mt < 2; ++mt) {
                const int rowLocal = mt * 16 + kh * 8;   // C layout: M = v + kh*8
                #pragma unroll
                for (int nt = 0; nt < 4; ++nt) {
                    const int ntg = ntBase + nt;
                    const int g   = ntg >> 1;
                    const int col = (ntg & 1) * 16 + lane15;
                    float* zp = &sm.z[br][g][rowLocal * 32 + col];
                    #pragma unroll
                    for (int v = 0; v < 8; ++v) zp[v * 32] = acc[mt][nt][v];
                }
            }
        }
        __syncthreads();
        for (int e = tid; e < 32 * 32; e += 256) {
            const int row = e >> 5;
            const int col = e & 31;
            const int gb  = rowBase + pass * 32 + row;
            const int gh  = colBase + col;
            const size_t oi = (size_t)gb * H_ + gh;
            float zf1 = sm.z[0][0][e] + bF1[gh];
            float zi1 = sm.z[0][1][e] + bI1[gh];
            float zc1 = sm.z[0][2][e] + bC1[gh];
            float zo1 = sm.z[0][3][e] + bO1[gh];
            float zf2 = sm.z[1][0][e] + bF2[gh];
            float zi2 = sm.z[1][1][e] + bI2[gh];
            float zc2 = sm.z[1][2][e] + bC2[gh];
            float zo2 = sm.z[1][3][e] + bO2[gh];
            float c0 = cprev[oi];
            float f1 = 1.0f / (1.0f + __expf(-zf1));
            float i1 = 1.0f / (1.0f + __expf(-zi1));
            float o1 = 1.0f / (1.0f + __expf(-zo1));
            float g1 = tanhf(zc1);
            float f2 = 1.0f / (1.0f + __expf(-zf2));
            float i2 = 1.0f / (1.0f + __expf(-zi2));
            float o2 = 1.0f / (1.0f + __expf(-zo2));
            float g2 = tanhf(zc2);
            float cn = (f1 * c0 + i1 * g1) + (f2 * c0 + i2 * g2);
            float hn = (o1 + o2) * tanhf(cn);
            outH[oi] = hn;
            outC[oi] = cn;
        }
    }
}

// ------------------------------- launcher -----------------------------------
extern "C" void kernel_launch(void* const* d_in, const int* in_sizes, int n_in,
                              void* d_out, int out_size, void* d_ws, size_t ws_size,
                              hipStream_t stream) {
    (void)in_sizes; (void)n_in; (void)out_size; (void)ws_size;
    const float* y  = (const float*)d_in[0];
    const float* hl = (const float*)d_in[1];
    const float* cl = (const float*)d_in[2];
    const float* ht = (const float*)d_in[3];
    // d_in[4] (c_temp) is unused by the reference.
    const float* w0 = (const float*)d_in[5];   const float* b0 = (const float*)d_in[6];
    const float* w1 = (const float*)d_in[7];   const float* b1 = (const float*)d_in[8];
    const float* w2 = (const float*)d_in[9];   const float* b2 = (const float*)d_in[10];
    const float* w3 = (const float*)d_in[11];  const float* b3 = (const float*)d_in[12];
    const float* w4 = (const float*)d_in[13];  const float* b4 = (const float*)d_in[14];
    const float* w5 = (const float*)d_in[15];  const float* b5 = (const float*)d_in[16];
    const float* w6 = (const float*)d_in[17];  const float* b6 = (const float*)d_in[18];
    const float* w7 = (const float*)d_in[19];  const float* b7 = (const float*)d_in[20];

    u16* hx = (u16*)d_ws;                        // 2*4096*1536 bf16 = 24 MB
    u16* Wb = hx + (size_t)2 * B_ * KD;          // 8*1024*1536 bf16 = 24 MB

    const int packBlocks = (int)((2L * B_ * (KD / 4) + 255) / 256);   // 12288
    pack_hx_kernel<<<packBlocks, 256, 0, stream>>>(y, hl, ht, hx);
    const int wBlocks = (int)((8L * H_ * (KD / 4) + 255) / 256);      // 12288
    pack_w_kernel<<<wBlocks, 256, 0, stream>>>(w0, w1, w2, w3, w4, w5, w6, w7, Wb);

    float* outH = (float*)d_out;
    float* outC = outH + (size_t)B_ * H_;
    lstm_main_kernel<<<dim3(B_ / BM, H_ / BN), 256, 0, stream>>>(
        hx, Wb, cl, b0, b1, b2, b3, b4, b5, b6, b7, outH, outC);
}
